// Mamba2TimeMix_85203561218348
// MI455X (gfx1250) — matrix operand order repaired and checked by reading, not compiled, not run on hardware
//
#include <hip/hip_runtime.h>
#include <stdint.h>

// ---------------- types ----------------
typedef uint32_t u32;
typedef uint64_t u64;
typedef unsigned short u16;
typedef __attribute__((ext_vector_type(16))) __bf16 v16bf;
typedef __attribute__((ext_vector_type(8)))  float  v8f;
typedef __attribute__((ext_vector_type(4)))  u32    v4u;
typedef __attribute__((ext_vector_type(8)))  int    v8i;
typedef __attribute__((ext_vector_type(4)))  int    v4i;
typedef __attribute__((ext_vector_type(4)))  u32    v4ui;
typedef __attribute__((ext_vector_type(8)))  u32    v8ui;

// ---------------- problem constants ----------------
// b=2, L=4096, d=1024, h=16, n=p=64, chunk=64 -> M=8192, chunks/seq=64
#define MROWS   8192
#define DMODEL  1024
#define NHEADS  16
#define NCHUNK  64
#define NBCH    2048        // b * chunks * heads

// ---------------- workspace byte offsets ----------------
// f32 buffers
#define BO_XIN   ((size_t)0)           // M*D f32; DEAD after xmul -> reused as sstates
#define BO_Z     ((size_t)33554432)    // M*D f32
#define BO_YD    ((size_t)67108864)    // M*D f32
#define BO_ST    ((size_t)100663296)   // NBCH*4096 f32 (chunk states)
#define BO_DT    ((size_t)134217728)   // M*H f32
#define BO_ACS   ((size_t)134742016)   // NBCH*64 f32
#define BO_ALAST ((size_t)135266304)   // NBCH f32
// bf16 buffers
#define BO_XBF   ((size_t)135274496)   // M*D bf16; DEAD after gemms -> reused as ybf
#define BO_XXBF  ((size_t)152051712)   // M*D bf16 (X = x_in * dt)
#define BO_BBF   ((size_t)168828928)   // M*D bf16
#define BO_CBF   ((size_t)185606144)   // M*D bf16
#define BO_WBF   ((size_t)202383360)   // W_in(2M) | W_B(1M) | W_C(1M) | W_out(1M) bf16
// total ~= 212.9 MB

// ---------------- WMMA helpers ----------------
__device__ __forceinline__ v8f wmma_bf16(v16bf a, v16bf b, v8f c) {
  return __builtin_amdgcn_wmma_f32_16x16x32_bf16(false, a, false, b,
                                                 (short)0, c, false, false);
}

// two 16B LDS loads -> one 32B WMMA operand
__device__ __forceinline__ v16bf frag_from2(const void* p0, const void* p1) {
  v4ui lo = *(const v4ui*)p0;
  v4ui hh = *(const v4ui*)p1;
  v8ui u;
  u[0] = lo[0]; u[1] = lo[1]; u[2] = lo[2]; u[3] = lo[3];
  u[4] = hh[0]; u[5] = hh[1]; u[6] = hh[2]; u[7] = hh[3];
  return __builtin_bit_cast(v16bf, u);
}

// A-matrix 16x32 bf16 fragment; tile row-major bf16, ld in elements.
// element e -> k = (e&7) + 16*(e>>3) + 8*hi : contiguous runs of 8.
__device__ __forceinline__ v16bf afrag_h(const u16* tile, int ld, int row,
                                         int k0, int hi) {
  const u16* r = tile + (size_t)row * ld + k0 + (hi << 3);
  return frag_from2(r, r + 16);
}

// B-matrix 32x16 bf16 fragment; Bmat[k, col] stored at tile[col*ld + k].
// lane col, element e -> k = k0 + (e&7) + 16*(e>>3) + 8*hi : contiguous runs of 8.
__device__ __forceinline__ v16bf bfrag_h(const u16* tile, int ld, int col,
                                         int k0, int hi) {
  const u16* r = tile + (size_t)col * ld + k0 + (hi << 3);
  return frag_from2(r, r + 16);
}

__device__ __forceinline__ void store_out(float* o, float v) { *o = v; }
__device__ __forceinline__ void store_out(__bf16* o, float v) { *o = (__bf16)v; }

// ---------------- TDM descriptor + issue ----------------
// flags word = D# group1 dword0 (data_size / pad config), see ISA 8.4
__device__ __forceinline__ void tdm_load_2d(u32 lds_off, const void* gptr,
                                            u32 tile_x, u32 tile_y, u64 stride,
                                            u32 flags) {
  u64 ga = (u64)(uintptr_t)gptr;
  v4u g0;
  g0[0] = 1u;                                   // count=1, no gather
  g0[1] = lds_off;                              // LDS byte address
  g0[2] = (u32)ga;                              // global_addr[31:0]
  g0[3] = (u32)((ga >> 32) & 0x01FFFFFFu) | (2u << 30);  // addr[56:32]|type=2
  u32 td0 = (u32)stride;                        // tensor_dim0 (elements)
  u32 td1 = 1u << 20;                           // tensor_dim1 (big, in-bounds)
  v8i g1;
  g1[0] = (int)flags;
  g1[1] = (int)((td0 & 0xFFFFu) << 16);
  g1[2] = (int)((td0 >> 16) | ((td1 & 0xFFFFu) << 16));
  g1[3] = (int)((td1 >> 16) | (tile_x << 16));
  g1[4] = (int)(tile_y & 0xFFFFu);
  g1[5] = (int)(u32)(stride & 0xFFFFFFFFu);
  g1[6] = (int)(u32)((stride >> 32) & 0xFFFFu);
  g1[7] = 0;
  v4i z4 = {0, 0, 0, 0};
  v8i z8 = {0, 0, 0, 0, 0, 0, 0, 0};
  __builtin_amdgcn_tensor_load_to_lds(g0, g1, z4, z4, z8, 0);
}
// bf16 tile: data_size=2B (code 1); pad 4 DWORDs after every 16 DWORDs
// (64B row -> 80B LDS row stride, 16B aligned, bank-spread)
#define TDM_FLAGS_BF16 ((1u << 16) | (1u << 20) | (3u << 22) | (3u << 25))

// ---------------- Kernel: tiled bf16-WMMA GEMM ----------------
// Out[m, j] = sum_k Ain[m, k] * W[j, k]   (bf16 row-major, f32 accum)
// 128 threads (4 waves), tile 64x64, K-step 32, TDM double-buffered.
#define GLDH  40            // padded LDS row stride (bf16 elements) = 80B
#define TILEH (64 * GLDH)   // one 64x32 bf16 tile (padded) = 2560 elements

template <typename OT>
__global__ __launch_bounds__(128) void gemm_kernel(
    const __bf16* __restrict__ Ain, const __bf16* __restrict__ W,
    OT* __restrict__ Out, int Kdim, int Ndim) {
  __shared__ __attribute__((aligned(16))) u16 lds[4 * TILEH];
  size_t m0 = (size_t)blockIdx.y * 64;
  size_t n0 = (size_t)blockIdx.x * 64;
  int tid = threadIdx.x;
  int wv = tid >> 5, lane = tid & 31;
  int hi = lane >> 4, l15 = lane & 15;
  v8f zero = {0.f, 0.f, 0.f, 0.f, 0.f, 0.f, 0.f, 0.f};
  v8f acc[4] = {zero, zero, zero, zero};
  int nK = Kdim >> 5;

  if (tid < 32) {  // wave 0 drives the TDM
    tdm_load_2d((u32)(uintptr_t)&lds[0], Ain + m0 * Kdim, 32, 64, (u64)Kdim,
                TDM_FLAGS_BF16);
    tdm_load_2d((u32)(uintptr_t)&lds[TILEH], W + n0 * Kdim, 32, 64, (u64)Kdim,
                TDM_FLAGS_BF16);
  }
  for (int i = 0; i < nK; ++i) {
    int cur = i & 1;
    if (tid < 32) {
      if (i + 1 < nK) {
        size_t kk = (size_t)(i + 1) << 5;
        tdm_load_2d((u32)(uintptr_t)&lds[(cur ^ 1) * 2 * TILEH],
                    Ain + m0 * Kdim + kk, 32, 64, (u64)Kdim, TDM_FLAGS_BF16);
        tdm_load_2d((u32)(uintptr_t)&lds[(cur ^ 1) * 2 * TILEH + TILEH],
                    W + n0 * Kdim + kk, 32, 64, (u64)Kdim, TDM_FLAGS_BF16);
        __builtin_amdgcn_s_wait_tensorcnt(2);  // current pair done
      } else {
        __builtin_amdgcn_s_wait_tensorcnt(0);
      }
    }
    __syncthreads();
    const u16* As = &lds[cur * 2 * TILEH];
    const u16* Ws = &lds[cur * 2 * TILEH + TILEH];
    v16bf a = afrag_h(As, GLDH, wv * 16 + l15, 0, hi);
#pragma unroll
    for (int nt = 0; nt < 4; ++nt) {
      v16bf b = bfrag_h(Ws, GLDH, nt * 16 + l15, 0, hi);
      acc[nt] = wmma_bf16(a, b, acc[nt]);
    }
    __syncthreads();
  }
#pragma unroll
  for (int nt = 0; nt < 4; ++nt)
#pragma unroll
    for (int r = 0; r < 8; ++r)
      store_out(&Out[(m0 + wv * 16 + r + hi * 8) * (size_t)Ndim + n0 +
                     nt * 16 + l15],
                acc[nt][r]);
}

// ---------------- Kernel: f32 -> bf16 convert ----------------
__global__ __launch_bounds__(256) void cvt_kernel(const float* __restrict__ in,
                                                  __bf16* __restrict__ out) {
  size_t i = (size_t)blockIdx.x * 256 + threadIdx.x;
  out[i] = (__bf16)in[i];
}

// ---------------- Kernel: dt = softplus(x @ W_dt.T + b_dt) ----------------
__global__ __launch_bounds__(256) void dt_kernel(const float* __restrict__ x,
                                                 const float* __restrict__ Wdt,
                                                 const float* __restrict__ bdt,
                                                 float* __restrict__ dt) {
  size_t m0 = (size_t)blockIdx.x * 16;
  int r = threadIdx.x >> 4, o = threadIdx.x & 15;
  const float* xr = x + (m0 + r) * DMODEL;
  const float* wr = Wdt + (size_t)o * DMODEL;
  float s = 0.f;
  for (int k = 0; k < DMODEL; ++k) s += xr[k] * wr[k];
  s += bdt[o];
  float sp = (s > 20.f) ? s : log1pf(__expf(s));
  dt[(m0 + r) * NHEADS + o] = sp;
}

// ---------------- Kernel: Xbf = bf16(x_in * dt) ----------------
__global__ __launch_bounds__(256) void xmul_kernel(
    const float* __restrict__ xin, const float* __restrict__ dt,
    __bf16* __restrict__ Xb) {
  size_t i = (size_t)blockIdx.x * 256 + threadIdx.x;
  size_t m = i >> 10;
  int h = (int)((i & 1023) >> 6);
  Xb[i] = (__bf16)(xin[i] * dt[m * NHEADS + h]);
}

// ---------------- Kernel: per-chunk cumsum of A ----------------
__global__ __launch_bounds__(64) void cumsum_kernel(
    const float* __restrict__ dt, const float* __restrict__ A_log,
    float* __restrict__ Acs, float* __restrict__ Alast) {
  __shared__ float s[64];
  int blk = blockIdx.x;
  int h = blk & 15, c = (blk >> 4) & 63, bb = blk >> 10;
  size_t m0 = (size_t)bb * 4096 + (size_t)c * 64;
  int t = threadIdx.x;
  float a = -__expf(A_log[h]) * dt[(m0 + t) * NHEADS + h];
  s[t] = a;
  __syncthreads();
  for (int off = 1; off < 64; off <<= 1) {
    float v = (t >= off) ? s[t - off] : 0.f;
    __syncthreads();
    s[t] += v;
    __syncthreads();
  }
  Acs[(size_t)blk * 64 + t] = s[t];
  if (t == 63) Alast[blk] = s[63];
}

// ---------------- Kernel: intra-chunk SSD ----------------
// one block per (b, chunk, head); 4 waves, 16-row stripes.  All LDS tiles
// bf16 with 72-element (144B) padded rows so every fragment is 2x b128.
#define CLD 72

__global__ __launch_bounds__(128) void ssd_chunk_kernel(
    const __bf16* __restrict__ Bb, const __bf16* __restrict__ Cb,
    const __bf16* __restrict__ Xb, const float* __restrict__ Acs,
    float* __restrict__ Ydiag, float* __restrict__ states) {
  __shared__ __attribute__((aligned(16))) __bf16 Cs[64 * CLD];  // C rows; -> M
  __shared__ __attribute__((aligned(16))) __bf16 Bs[64 * CLD];  // B rows
  __shared__ __attribute__((aligned(16))) __bf16 Bt[64 * CLD];  // (B*dec)^T
  __shared__ __attribute__((aligned(16))) __bf16 Xt[64 * CLD];  // X^T (p-major)
  __shared__ float As[64];
  int blk = blockIdx.x;
  int h = blk & 15, c = (blk >> 4) & 63, bb = blk >> 10;
  size_t m0 = (size_t)bb * 4096 + (size_t)c * 64;
  int hc = h * 64;
  int tid = threadIdx.x;
  for (int i = tid; i < 2048; i += 128) {
    int r = i >> 5, c2 = (i & 31) << 1;
    size_t g = (m0 + r) * DMODEL + hc + c2;
    *(u32*)&Cs[r * CLD + c2] = *(const u32*)&Cb[g];
    *(u32*)&Bs[r * CLD + c2] = *(const u32*)&Bb[g];
    __bf16 x0 = Xb[g], x1 = Xb[g + 1];
    Xt[c2 * CLD + r] = x0;
    Xt[(c2 + 1) * CLD + r] = x1;
  }
  if (tid < 64) As[tid] = Acs[(size_t)blk * 64 + tid];
  __syncthreads();
  float alast = As[63];
  // decayed transpose of B (A-operand of the states matmul)
  for (int i = tid; i < 4096; i += 128) {
    int n = i >> 6, l = i & 63;
    Bt[n * CLD + l] = (__bf16)((float)Bs[l * CLD + n] * __expf(alast - As[l]));
  }
  __syncthreads();

  int wv = tid >> 5, lane = tid & 31;
  int hi = lane >> 4, l15 = lane & 15;
  v8f zero = {0.f, 0.f, 0.f, 0.f, 0.f, 0.f, 0.f, 0.f};

  // phase 1: G = C @ B^T  (K = n)
  v8f g[4] = {zero, zero, zero, zero};
#pragma unroll
  for (int k0 = 0; k0 < 64; k0 += 32) {
    v16bf a = afrag_h((const u16*)Cs, CLD, wv * 16 + l15, k0, hi);
#pragma unroll
    for (int st = 0; st < 4; ++st) {
      v16bf b = bfrag_h((const u16*)Bs, CLD, st * 16 + l15, k0, hi);
      g[st] = wmma_bf16(a, b, g[st]);
    }
  }
  // mask + segsum decay -> M written into Cs (each wave owns its rows)
#pragma unroll
  for (int st = 0; st < 4; ++st) {
    int s = st * 16 + l15;
#pragma unroll
    for (int r = 0; r < 8; ++r) {
      int l = wv * 16 + r + hi * 8;
      float v = (s <= l) ? g[st][r] * __expf(As[l] - As[s]) : 0.f;
      Cs[l * CLD + s] = (__bf16)v;
    }
  }
  __syncthreads();

  // phase 2: Y_diag = M @ X  (K = s); B-operand rows from Xt
  v8f yd[4] = {zero, zero, zero, zero};
#pragma unroll
  for (int k0 = 0; k0 < 64; k0 += 32) {
    v16bf a = afrag_h((const u16*)Cs, CLD, wv * 16 + l15, k0, hi);
#pragma unroll
    for (int pt = 0; pt < 4; ++pt) {
      v16bf b = bfrag_h((const u16*)Xt, CLD, pt * 16 + l15, k0, hi);
      yd[pt] = wmma_bf16(a, b, yd[pt]);
    }
  }
#pragma unroll
  for (int pt = 0; pt < 4; ++pt)
#pragma unroll
    for (int r = 0; r < 8; ++r) {
      int l = wv * 16 + r + hi * 8;
      Ydiag[(m0 + l) * DMODEL + hc + pt * 16 + l15] = yd[pt][r];
    }

  // phase 3: states[n, p] = sum_l Bt[n, l] * X[l, p]
  v8f stc[4] = {zero, zero, zero, zero};
#pragma unroll
  for (int k0 = 0; k0 < 64; k0 += 32) {
    v16bf a = afrag_h((const u16*)Bt, CLD, wv * 16 + l15, k0, hi);
#pragma unroll
    for (int pt = 0; pt < 4; ++pt) {
      v16bf b = bfrag_h((const u16*)Xt, CLD, pt * 16 + l15, k0, hi);
      stc[pt] = wmma_bf16(a, b, stc[pt]);
    }
  }
#pragma unroll
  for (int pt = 0; pt < 4; ++pt)
#pragma unroll
    for (int r = 0; r < 8; ++r)
      states[(size_t)blk * 4096 + (wv * 16 + r + hi * 8) * 64 + pt * 16 + l15] =
          stc[pt][r];
}

// ---------------- Kernel: inter-chunk state scan ----------------
__global__ __launch_bounds__(256) void scan_kernel(
    const float* __restrict__ states, const float* __restrict__ Alast,
    float* __restrict__ sstates, float* __restrict__ fstate) {
  int bh = blockIdx.x;
  int bb = bh >> 4, h = bh & 15;
  int tid = threadIdx.x;
  float run[16];
#pragma unroll
  for (int j = 0; j < 16; ++j) run[j] = 0.f;
  for (int c = 0; c < NCHUNK; ++c) {
    int blk = (bb * 64 + c) * 16 + h;
    size_t base = (size_t)blk * 4096;
    float ad = __expf(Alast[blk]);
#pragma unroll
    for (int j = 0; j < 16; ++j) {
      int e = tid + j * 256;
      float st = states[base + e];
      sstates[base + e] = run[j];
      run[j] = run[j] * ad + st;
    }
  }
  size_t fb = (size_t)bh * 4096;
#pragma unroll
  for (int j = 0; j < 16; ++j) {
    int e = tid + j * 256;
    int n = e >> 6, p = e & 63;
    fstate[fb + (size_t)p * 64 + n] = run[j];  // (b, h, p, n)
  }
}

// ---------------- Kernel: Y_off + gated epilogue -> bf16 ----------------
__global__ __launch_bounds__(128) void yoff_kernel(
    const __bf16* __restrict__ Cb, const float* __restrict__ SS,
    const float* __restrict__ Acs, const float* __restrict__ zb,
    const float* __restrict__ Yd, __bf16* __restrict__ yout) {
  __shared__ __attribute__((aligned(16))) __bf16 Cs[64 * CLD];
  __shared__ __attribute__((aligned(16))) __bf16 St[64 * CLD];  // S^T (p-major)
  __shared__ float As[64];
  int blk = blockIdx.x;
  int h = blk & 15, c = (blk >> 4) & 63, bb = blk >> 10;
  size_t m0 = (size_t)bb * 4096 + (size_t)c * 64;
  int hc = h * 64;
  int tid = threadIdx.x;
  for (int i = tid; i < 2048; i += 128) {
    int r = i >> 5, c2 = (i & 31) << 1;
    *(u32*)&Cs[r * CLD + c2] = *(const u32*)&Cb[(m0 + r) * DMODEL + hc + c2];
  }
  for (int i = tid; i < 4096; i += 128) {
    int n = i >> 6, p = i & 63;
    St[p * CLD + n] = (__bf16)SS[(size_t)blk * 4096 + i];  // SS is (n, p)
  }
  if (tid < 64) As[tid] = Acs[(size_t)blk * 64 + tid];
  __syncthreads();

  int wv = tid >> 5, lane = tid & 31;
  int hi = lane >> 4, l15 = lane & 15;
  v8f zero = {0.f, 0.f, 0.f, 0.f, 0.f, 0.f, 0.f, 0.f};
  v8f acc[4] = {zero, zero, zero, zero};
#pragma unroll
  for (int k0 = 0; k0 < 64; k0 += 32) {
    v16bf a = afrag_h((const u16*)Cs, CLD, wv * 16 + l15, k0, hi);
#pragma unroll
    for (int pt = 0; pt < 4; ++pt) {
      v16bf b = bfrag_h((const u16*)St, CLD, pt * 16 + l15, k0, hi);
      acc[pt] = wmma_bf16(a, b, acc[pt]);
    }
  }
#pragma unroll
  for (int pt = 0; pt < 4; ++pt)
#pragma unroll
    for (int r = 0; r < 8; ++r) {
      int l = wv * 16 + r + hi * 8;
      size_t gi = (m0 + l) * DMODEL + hc + pt * 16 + l15;
      float y = Yd[gi] + acc[pt][r] * __expf(As[l]);
      float zv = zb[gi];
      y *= zv / (1.f + __expf(-zv));  // silu gate
      yout[gi] = (__bf16)y;
    }
}

// ---------------- launcher ----------------
extern "C" void kernel_launch(void* const* d_in, const int* in_sizes, int n_in,
                              void* d_out, int out_size, void* d_ws,
                              size_t ws_size, hipStream_t stream) {
  (void)in_sizes; (void)n_in; (void)out_size; (void)ws_size;
  const float* x     = (const float*)d_in[0];
  const float* W_in  = (const float*)d_in[1];
  const float* W_dt  = (const float*)d_in[2];
  const float* b_dt  = (const float*)d_in[3];
  const float* W_B   = (const float*)d_in[4];
  const float* W_C   = (const float*)d_in[5];
  const float* W_out = (const float*)d_in[6];
  const float* A_log = (const float*)d_in[7];
  float* out = (float*)d_out;
  char* ws = (char*)d_ws;

  float*  xin   = (float*)(ws + BO_XIN);
  float*  zb    = (float*)(ws + BO_Z);
  float*  Yd    = (float*)(ws + BO_YD);
  float*  Stt   = (float*)(ws + BO_ST);
  float*  dtb   = (float*)(ws + BO_DT);
  float*  Acs   = (float*)(ws + BO_ACS);
  float*  Alast = (float*)(ws + BO_ALAST);
  __bf16* xbf   = (__bf16*)(ws + BO_XBF);
  __bf16* Xbf   = (__bf16*)(ws + BO_XXBF);
  __bf16* Bbf   = (__bf16*)(ws + BO_BBF);
  __bf16* Cbf   = (__bf16*)(ws + BO_CBF);
  __bf16* wbf_in  = (__bf16*)(ws + BO_WBF);
  __bf16* wbf_B   = wbf_in + (size_t)2 * DMODEL * DMODEL;
  __bf16* wbf_C   = wbf_B + (size_t)DMODEL * DMODEL;
  __bf16* wbf_out = wbf_C + (size_t)DMODEL * DMODEL;
  float*  SSt   = xin;   // sstates reuses dead x_in f32
  __bf16* ybf   = xbf;   // gated y reuses dead x_bf

  dim3 gblk(128);
  dim3 ggrid(DMODEL / 64, MROWS / 64);  // (16, 128)

  // bf16 conversions (one pass each)
  cvt_kernel<<<(MROWS * DMODEL) / 256, 256, 0, stream>>>(x, xbf);
  cvt_kernel<<<(2 * DMODEL * DMODEL) / 256, 256, 0, stream>>>(W_in, wbf_in);
  cvt_kernel<<<(DMODEL * DMODEL) / 256, 256, 0, stream>>>(W_B, wbf_B);
  cvt_kernel<<<(DMODEL * DMODEL) / 256, 256, 0, stream>>>(W_C, wbf_C);
  cvt_kernel<<<(DMODEL * DMODEL) / 256, 256, 0, stream>>>(W_out, wbf_out);

  // projections (bf16 WMMA GEMMs, TDM-staged)
  gemm_kernel<float><<<ggrid, gblk, 0, stream>>>(xbf, wbf_in, xin, DMODEL,
                                                 DMODEL);
  gemm_kernel<float><<<ggrid, gblk, 0, stream>>>(
      xbf, wbf_in + (size_t)DMODEL * DMODEL, zb, DMODEL, DMODEL);
  gemm_kernel<__bf16><<<ggrid, gblk, 0, stream>>>(xbf, wbf_B, Bbf, DMODEL,
                                                  DMODEL);
  gemm_kernel<__bf16><<<ggrid, gblk, 0, stream>>>(xbf, wbf_C, Cbf, DMODEL,
                                                  DMODEL);

  dt_kernel<<<MROWS / 16, 256, 0, stream>>>(x, W_dt, b_dt, dtb);
  xmul_kernel<<<(MROWS * DMODEL) / 256, 256, 0, stream>>>(xin, dtb, Xbf);
  cumsum_kernel<<<NBCH, 64, 0, stream>>>(dtb, A_log, Acs, Alast);

  ssd_chunk_kernel<<<NBCH, 128, 0, stream>>>(Bbf, Cbf, Xbf, Acs, Yd, Stt);
  scan_kernel<<<32, 256, 0, stream>>>(Stt, Alast, SSt,
                                      out + (size_t)MROWS * DMODEL);
  yoff_kernel<<<NBCH, 128, 0, stream>>>(Cbf, SSt, Acs, zb, Yd, ybf);

  // output projection
  gemm_kernel<float><<<ggrid, gblk, 0, stream>>>(ybf, wbf_out, out, DMODEL,
                                                 DMODEL);
}
